// MAM_48601849921883
// MI455X (gfx1250) — matrix operand order repaired and checked by reading, not compiled, not run on hardware
//
#include <hip/hip_runtime.h>

// MI455X / gfx1250: wave32, WMMA 16x16x32 bf16 -> f32.
// Flash-attention structure; K/V tiles double-buffered in LDS via
// GLOBAL_LOAD_ASYNC_TO_LDS_B128 (ASYNCcnt), coalesced float4 epilogue.

typedef __attribute__((ext_vector_type(16))) __bf16 v16bf;
typedef __attribute__((ext_vector_type(8)))  float  v8f;

union F16 {
  v16bf v;
  uint4 q[2];
};

constexpr int kB = 8;     // batch
constexpr int kC = 256;   // Cp = 64 * PATCH^2
constexpr int kN = 4096;  // Hp * Wp

// LDS layout (bytes) for attn kernel, dynamic shared memory:
//   Kt[2][32][264] bf16 : 0      .. 33791   (row = 528B)
//   Vt[2][256][40] bf16 : 33792  .. 74751   (row = 80B)
//   Pl[8][16][40]  bf16 : 74752  .. 84991
//   epilogue reuse: Os[128][133] f32 (68096B) aliases from offset 0
constexpr unsigned kKtRow   = 264;
constexpr unsigned kVtRow   = 40;
constexpr unsigned kKtBytes = 32 * kKtRow * 2;    // 16896
constexpr unsigned kVtBytes = 256 * kVtRow * 2;   // 20480
constexpr unsigned kVtBase  = 2 * kKtBytes;       // 33792
constexpr unsigned kPlBase  = kVtBase + 2 * kVtBytes;  // 74752
constexpr unsigned kSmemBytes = kPlBase + 8 * 16 * kVtRow * 2;  // 84992

__device__ inline v8f wmma_bf16(v16bf a, v16bf b, v8f c) {
  return __builtin_amdgcn_wmma_f32_16x16x32_bf16(false, a, false, b, (short)0, c,
                                                 false, false);
}

__device__ inline v16bf cvt16f(float4 a, float4 b, float4 c, float4 d) {
  v16bf r;
  r[0]  = (__bf16)a.x; r[1]  = (__bf16)a.y; r[2]  = (__bf16)a.z; r[3]  = (__bf16)a.w;
  r[4]  = (__bf16)b.x; r[5]  = (__bf16)b.y; r[6]  = (__bf16)b.z; r[7]  = (__bf16)b.w;
  r[8]  = (__bf16)c.x; r[9]  = (__bf16)c.y; r[10] = (__bf16)c.z; r[11] = (__bf16)c.w;
  r[12] = (__bf16)d.x; r[13] = (__bf16)d.y; r[14] = (__bf16)d.z; r[15] = (__bf16)d.w;
  return r;
}

// --- CDNA5 async memory->LDS (tracked by ASYNCcnt) ---
__device__ inline void async_b128(unsigned lds_addr, const void* gaddr) {
  asm volatile("global_load_async_to_lds_b128 %0, %1, off"
               :: "v"(lds_addr), "v"(gaddr)
               : "memory");
}

__device__ inline void wait_async0() {
#if __has_builtin(__builtin_amdgcn_s_wait_asynccnt)
  __builtin_amdgcn_s_wait_asynccnt(0);
#else
  asm volatile("s_wait_asynccnt 0x0" ::: "memory");
#endif
}

// ---------------------------------------------------------------------------
// Kernel 1: QKV projections (~9% of total FLOPs).
//   z=0: Q -> QT[b][n][c]  z=1: K -> KT[b][n][c]  z=2: V -> Vn[b][c][m]
// ---------------------------------------------------------------------------
__global__ __launch_bounds__(256) void qkv_kernel(
    const float* __restrict__ radar, const float* __restrict__ modal,
    const float* __restrict__ Wq, const float* __restrict__ bq,
    const float* __restrict__ Wk, const float* __restrict__ bk,
    const float* __restrict__ Wv, const float* __restrict__ bv,
    const float* __restrict__ pos_r, const float* __restrict__ pos_m,
    __bf16* __restrict__ QT, __bf16* __restrict__ KT, __bf16* __restrict__ Vn)
{
  const int tid  = threadIdx.x;
  const int wave = tid >> 5;
  const int lane = tid & 31;
  const int lr   = lane & 15;
  const int hi   = lane >> 4;
  const int n0   = blockIdx.x * 64;
  const int b    = blockIdx.y;
  const int z    = blockIdx.z;

  const float* x    = (z == 0) ? radar : modal;
  const float* pos  = (z == 0) ? pos_r : pos_m;
  const float* W    = (z == 0) ? Wq : (z == 1 ? Wk : Wv);
  const float* bias = (z == 0) ? bq : (z == 1 ? bk : bv);

  __shared__ __bf16 xT[64][40];  // [n][c] tile, padded row = 80B

  const v8f zero8 = {0.f, 0.f, 0.f, 0.f, 0.f, 0.f, 0.f, 0.f};
  v8f acc[8];
  for (int i = 0; i < 8; ++i) acc[i] = zero8;

  for (int ks = 0; ks < 8; ++ks) {
    const int c0 = ks * 32;
    __syncthreads();
    for (int i = tid; i < 32 * 64; i += 256) {
      const int c = i >> 6;
      const int n = i & 63;
      const size_t gi = (size_t)(c0 + c) * kN + (n0 + n);
      xT[n][c] = (__bf16)(x[(size_t)b * kC * kN + gi] + pos[gi]);
    }
    __syncthreads();

    if (z < 2) {
      const int nt = (wave & 3) * 16;
      const int ob = (wave >> 2) * 128;
      F16 af;
      const __bf16* ar = &xT[nt + lr][0];
      af.q[0] = *(const uint4*)(ar + hi * 16);
      af.q[1] = *(const uint4*)(ar + hi * 16 + 8);
      for (int ot = 0; ot < 8; ++ot) {
        const float4* w4 =
            (const float4*)(W + (size_t)(ob + ot * 16 + lr) * kC + c0 + hi * 16);
        F16 bfr;
        bfr.v = cvt16f(w4[0], w4[1], w4[2], w4[3]);
        acc[ot] = wmma_bf16(af.v, bfr.v, acc[ot]);
      }
    } else {
      for (int mt = 0; mt < 2; ++mt) {
        const float* wr = W + (size_t)(wave * 32 + mt * 16 + lr) * kC + c0;
        float4 a0 = *(const float4*)(wr + hi * 8);
        float4 a1 = *(const float4*)(wr + hi * 8 + 4);
        float4 a2 = *(const float4*)(wr + 16 + hi * 8);
        float4 a3 = *(const float4*)(wr + 16 + hi * 8 + 4);
        F16 af;
        af.v = cvt16f(a0, a1, a2, a3);
        for (int nt = 0; nt < 4; ++nt) {
          F16 bfr;
          const __bf16* xr = &xT[nt * 16 + lr][hi * 8];
          bfr.q[0] = *(const uint4*)xr;
          bfr.q[1] = *(const uint4*)(xr + 16);
          acc[mt * 4 + nt] = wmma_bf16(af.v, bfr.v, acc[mt * 4 + nt]);
        }
      }
    }
  }

  if (z < 2) {
    __bf16* out = (z == 0) ? QT : KT;
    const int nt = (wave & 3) * 16;
    const int ob = (wave >> 2) * 128;
    for (int ot = 0; ot < 8; ++ot) {
      const int o = ob + ot * 16 + lr;
      const float bb = bias[o];
      for (int r = 0; r < 8; ++r) {
        const int n = n0 + nt + r + hi * 8;
        out[((size_t)b * kN + n) * kC + o] = (__bf16)(acc[ot][r] + bb);
      }
    }
  } else {
    for (int mt = 0; mt < 2; ++mt) {
      for (int r = 0; r < 8; ++r) {
        const int c = wave * 32 + mt * 16 + r + hi * 8;
        const float bb = bias[c];
        for (int nt = 0; nt < 4; ++nt) {
          const int n = n0 + nt * 16 + lr;
          Vn[((size_t)b * kC + c) * kN + n] = (__bf16)(acc[mt * 4 + nt][r] + bb);
        }
      }
    }
  }
}

// ---------------------------------------------------------------------------
// Kernel 2: flash attention with async double-buffered K/V tiles.
// Block = 8 waves, 128 query rows (16/wave), full 256-wide f32 accumulator.
// ---------------------------------------------------------------------------
__global__ __launch_bounds__(256) void attn_kernel(
    const __bf16* __restrict__ QT, const __bf16* __restrict__ KT,
    const __bf16* __restrict__ Vn,
    const float* __restrict__ radar, const float* __restrict__ modal,
    float* __restrict__ out)
{
  extern __shared__ char smem_raw[];
  const int tid  = threadIdx.x;
  const int wave = tid >> 5;
  const int lane = tid & 31;
  const int lr   = lane & 15;
  const int hi   = lane >> 4;
  const int b    = blockIdx.y;
  const int n0   = blockIdx.x * 128;
  const int nw   = n0 + wave * 16;

  // runtime-computed LDS addresses (no aggregate initializers from LDS symbol)
  const unsigned smbase = (unsigned)(size_t)smem_raw;
  __bf16* Plw = (__bf16*)(smem_raw + kPlBase) + wave * 16 * kVtRow;

  // per-thread async staging addresses (K: 8 threads/row, V: 1 thread/row)
  const int krow = tid >> 3;
  const int kco  = (tid & 7) * 32;
  const __bf16* kg0 = KT + ((size_t)b * kN + krow) * kC + kco;
  const __bf16* vg0 = Vn + ((size_t)b * kC + tid) * kN;
  const unsigned klA = smbase + (krow * kKtRow + kco) * 2;
  const unsigned vlA = smbase + kVtBase + tid * kVtRow * 2;

  // Q fragments for this wave's 16 rows, resident in VGPRs
  v16bf qf[8];
  {
    const __bf16* qr = QT + ((size_t)b * kN + nw + lr) * kC;
    for (int k = 0; k < 8; ++k) {
      F16 f;
      f.q[0] = *(const uint4*)(qr + k * 32 + hi * 16);
      f.q[1] = *(const uint4*)(qr + k * 32 + hi * 16 + 8);
      qf[k] = f.v;
    }
  }

  const v8f zero8 = {0.f, 0.f, 0.f, 0.f, 0.f, 0.f, 0.f, 0.f};
  v8f oacc[16];
  for (int i = 0; i < 16; ++i) oacc[i] = zero8;
  float mrow[8], lrow[8];
  for (int r = 0; r < 8; ++r) { mrow[r] = -3.0e38f; lrow[r] = 0.f; }

  // prologue: async-stage step 0 into buffer 0
  for (int j = 0; j < 4; ++j) async_b128(klA + j * 16, kg0 + j * 8);
  for (int j = 0; j < 4; ++j) async_b128(vlA + j * 16, vg0 + j * 8);
  wait_async0();
  __syncthreads();

  for (int step = 0; step < kN / 32; ++step) {
    const int m0  = step * 32;
    const int cur = step & 1;

    // prefetch next m-step into the other buffer (its last readers all
    // passed the barrier that ended step-1)
    if (step + 1 < kN / 32) {
      const __bf16* kg = kg0 + (size_t)(m0 + 32) * kC;
      const __bf16* vg = vg0 + (m0 + 32);
      const unsigned nb = 1u - (unsigned)cur;
      const unsigned kl = klA + nb * kKtBytes;
      const unsigned vl = vlA + nb * kVtBytes;
      for (int j = 0; j < 4; ++j) async_b128(kl + j * 16, kg + j * 8);
      for (int j = 0; j < 4; ++j) async_b128(vl + j * 16, vg + j * 8);
    }

    const __bf16* Kc = (const __bf16*)(smem_raw + cur * kKtBytes);
    const __bf16* Vc = (const __bf16*)(smem_raw + kVtBase + cur * kVtBytes);

    // S = Q^T K (two 16-wide m tiles, reduce over c = 8 x 32)
    v8f s0 = zero8, s1 = zero8;
    for (int k = 0; k < 8; ++k) {
      F16 b0, b1;
      const __bf16* k0 = Kc + lr * kKtRow + k * 32 + hi * 16;
      b0.q[0] = *(const uint4*)k0;
      b0.q[1] = *(const uint4*)(k0 + 8);
      const __bf16* k1 = Kc + (16 + lr) * kKtRow + k * 32 + hi * 16;
      b1.q[0] = *(const uint4*)k1;
      b1.q[1] = *(const uint4*)(k1 + 8);
      s0 = wmma_bf16(qf[k], b0.v, s0);
      s1 = wmma_bf16(qf[k], b1.v, s1);
    }

    // online softmax; D layout: VGPR r = row (r + hi*8), 16 lanes = 16 m's
    float oscale[8];
    for (int r = 0; r < 8; ++r) {
      float mx = fmaxf(s0[r], s1[r]);
      for (int d = 1; d < 16; d <<= 1) mx = fmaxf(mx, __shfl_xor(mx, d, 32));
      const float mnew  = fmaxf(mrow[r], mx);
      const float scale = __expf(mrow[r] - mnew);
      const float p0 = __expf(s0[r] - mnew);
      const float p1 = __expf(s1[r] - mnew);
      s0[r] = p0;
      s1[r] = p1;
      float ps = p0 + p1;
      for (int d = 1; d < 16; d <<= 1) ps += __shfl_xor(ps, d, 32);
      lrow[r]   = lrow[r] * scale + ps;
      mrow[r]   = mnew;
      oscale[r] = scale;
    }
    for (int ct = 0; ct < 16; ++ct)
      for (int r = 0; r < 8; ++r) oacc[ct][r] *= oscale[r];

    // P: D layout -> A layout through wave-private LDS patch (DS in-order)
    for (int r = 0; r < 8; ++r) {
      Plw[(r + hi * 8) * kVtRow + lr]      = (__bf16)s0[r];
      Plw[(r + hi * 8) * kVtRow + 16 + lr] = (__bf16)s1[r];
    }
    F16 pf;
    pf.q[0] = *(const uint4*)(Plw + lr * kVtRow + hi * 16);
    pf.q[1] = *(const uint4*)(Plw + lr * kVtRow + hi * 16 + 8);

    // O += P * V  (16 c-tiles, K = 32 m's)
    for (int ct = 0; ct < 16; ++ct) {
      F16 vf;
      const __bf16* vr = Vc + (ct * 16 + lr) * kVtRow + hi * 16;
      vf.q[0] = *(const uint4*)vr;
      vf.q[1] = *(const uint4*)(vr + 8);
      oacc[ct] = wmma_bf16(pf.v, vf.v, oacc[ct]);
    }

    if (step + 1 < kN / 32) wait_async0();
    __syncthreads();
  }

  // epilogue: normalize + residual, coalesced through LDS (Os[n][c], pad 133)
  float inv[8];
  for (int r = 0; r < 8; ++r) inv[r] = 1.0f / lrow[r];
  float* Os = (float*)smem_raw;  // [128][133] f32, aliases K/V buffers
  const size_t base = (size_t)b * kC * kN;

  for (int p = 0; p < 2; ++p) {
    __syncthreads();
    // scatter: lanes write consecutive c -> conflict-free
    for (int ct = p * 8; ct < p * 8 + 8; ++ct) {
      const int c = ct * 16 + lr - p * 128;
      for (int r = 0; r < 8; ++r)
        Os[(wave * 16 + r + hi * 8) * 133 + c] = oacc[ct][r] * inv[r];
    }
    __syncthreads();
    // flush: float4-coalesced along n, fused residual
    for (int i = tid; i < 128 * 32; i += 256) {
      const int c  = i >> 5;
      const int nq = (i & 31) * 4;
      const size_t idx = base + (size_t)(p * 128 + c) * kN + n0 + nq;
      float4 rr = *(const float4*)(radar + idx);
      float4 mm = *(const float4*)(modal + idx);
      float4 o;
      o.x = Os[(nq + 0) * 133 + c] + rr.x + mm.x;
      o.y = Os[(nq + 1) * 133 + c] + rr.y + mm.y;
      o.z = Os[(nq + 2) * 133 + c] + rr.z + mm.z;
      o.w = Os[(nq + 3) * 133 + c] + rr.w + mm.w;
      *(float4*)(out + idx) = o;
    }
  }
}

// ---------------------------------------------------------------------------
extern "C" void kernel_launch(void* const* d_in, const int* in_sizes, int n_in,
                              void* d_out, int out_size, void* d_ws, size_t ws_size,
                              hipStream_t stream) {
  const float* radar = (const float*)d_in[0];
  const float* modal = (const float*)d_in[1];
  const float* Wq    = (const float*)d_in[2];
  const float* bq    = (const float*)d_in[3];
  const float* Wk    = (const float*)d_in[4];
  const float* bk    = (const float*)d_in[5];
  const float* Wv    = (const float*)d_in[6];
  const float* bv    = (const float*)d_in[7];
  const float* pos_r = (const float*)d_in[8];
  const float* pos_m = (const float*)d_in[9];

  __bf16* QT = (__bf16*)d_ws;
  __bf16* KT = QT + (size_t)kB * kN * kC;
  __bf16* Vn = KT + (size_t)kB * kN * kC;

  dim3 g1(kN / 64, kB, 3);
  qkv_kernel<<<g1, 256, 0, stream>>>(radar, modal, Wq, bq, Wk, bk, Wv, bv,
                                     pos_r, pos_m, QT, KT, Vn);
  dim3 g2(kN / 128, kB);
  attn_kernel<<<g2, 256, kSmemBytes, stream>>>(QT, KT, Vn, radar, modal,
                                               (float*)d_out);
}
